// QMeasureDensityMPS_57509612094205
// MI455X (gfx1250) — matrix operand order repaired and checked by reading, not compiled, not run on hardware
//
#include <hip/hip_runtime.h>
#include <stdint.h>

// ---------------------------------------------------------------------------
// Reference: two GEMMs [B x 2046] @ [2046 x 64] over contiguous slices of each
// 4096-float x row, plus an 8x8 bilinear epilogue per sample.
//   Left  GEMM K padded to 2048 with TWO LEADING zero rows  -> A slice starts
//         at x + k*4096 (16B aligned); K'>=2 maps to site s=(K'-2)/2.
//   Right GEMM K padded to 2048 with two trailing zero rows -> A slice starts
//         at x + k*4096 + 2048 (16B aligned).
// Workspace layout (bytes):
//   [0   ..  64)  float lbn[16]  : left_b with row 2 normalized
//   [64  .. 128)  float rbn[16]  : right_b with row 2 normalized
//   [128 .. 136)  float scaleL, scaleR  (1/norm of left[3], right[3])
//   [256 .. 256+256K)        uint32 WL[65536]  bf16-pair swizzled fragments
//   [256+256K .. 256+512K)   uint32 WR[65536]
// ---------------------------------------------------------------------------

typedef __attribute__((ext_vector_type(16))) __bf16 v16bf;
typedef __attribute__((ext_vector_type(8)))  float  v8f;

__device__ __forceinline__ uint32_t bf16_rne_bits(float f) {
  union { float f; uint32_t u; } c; c.f = f;
  uint32_t u = c.u;
  return (u + 0x7FFFu + ((u >> 16) & 1u)) >> 16;   // round-to-nearest-even
}

// ------------------------- prep: norms + boundaries -------------------------
__global__ void mps_prep0(const float* __restrict__ left_b,
                          const float* __restrict__ left,
                          const float* __restrict__ right,
                          const float* __restrict__ right_b,
                          float* __restrict__ wsf) {
  const int t = threadIdx.x;
  if (t == 0) {                               // left_b, normalize row 2 (elems 4,5)
    float inv = 1.0f / sqrtf(left_b[4]*left_b[4] + left_b[5]*left_b[5]);
    for (int i = 0; i < 16; ++i) wsf[i] = left_b[i];
    wsf[4] *= inv; wsf[5] *= inv;
  } else if (t == 1) {                        // right_b, normalize row 2
    float inv = 1.0f / sqrtf(right_b[4]*right_b[4] + right_b[5]*right_b[5]);
    for (int i = 0; i < 16; ++i) wsf[16 + i] = right_b[i];
    wsf[20] *= inv; wsf[21] *= inv;
  } else if (t == 2) {                        // 1/frobenius(left[3]) : 128 elems @ 384
    float s = 0.f;
    for (int i = 0; i < 128; ++i) { float v = left[384 + i]; s += v*v; }
    wsf[32] = 1.0f / sqrtf(s);
  } else if (t == 3) {                        // 1/frobenius(right[3])
    float s = 0.f;
    for (int i = 0; i < 128; ++i) { float v = right[384 + i]; s += v*v; }
    wsf[33] = 1.0f / sqrtf(s);
  }
}

// ---------------- prep: weights -> pre-swizzled bf16 B fragments ------------
// Fragment dword index: e = ((kb*4 + nt)*32 + lane)*8 + v,  kb in [0,64),
// nt in [0,4), lane in [0,32), v = VGPR in [0,8).  Each dword = 2 bf16 (h=0,1).
// 16-bit B 32x16 layout (matches A table order): lane group g = lane>>4
// holds K = 8*g + 8*(v>>2) + 2*v + h;  N = nt*16 + (lane&15) = 8*a + b.
__global__ void mps_build_w(const float* __restrict__ left,
                            const float* __restrict__ right,
                            const float* __restrict__ wsf,
                            uint32_t* __restrict__ WL,
                            uint32_t* __restrict__ WR) {
  const int idx   = blockIdx.x * blockDim.x + threadIdx.x;  // 0 .. 131071
  const int which = idx >> 16;                               // 0 = L, 1 = R
  const int e     = idx & 0xFFFF;
  const int v    = e & 7;
  const int lane = (e >> 3) & 31;
  const int nt   = (e >> 8) & 3;
  const int kb   = e >> 10;                                  // 0..63
  const int g    = lane >> 4;
  const int n    = nt * 16 + (lane & 15);
  const int a    = n >> 3, b = n & 7;

  uint32_t outv = 0;
  #pragma unroll
  for (int h = 0; h < 2; ++h) {
    const int K = kb * 32 + (g << 3) + ((v >> 2) << 3) + 2 * v + h; // padded K in [0,2048)
    float val = 0.f;
    if (which == 0) {
      if (K >= 2) {                                          // two leading zero rows
        const int kk = K - 2, s = kk >> 1, p = kk & 1;       // s in [0,1023) always
        val = left[((s * 8 + a) * 8 + b) * 2 + p];
        if (s == 3) val *= wsf[32];
      }
    } else {
      if (K < 2046) {                                        // two trailing zero rows
        const int s = K >> 1, p = K & 1;
        val = right[((s * 8 + a) * 8 + b) * 2 + p];
        if (s == 3) val *= wsf[33];
      }
    }
    outv |= bf16_rne_bits(val) << (16 * h);
  }
  (which ? WR : WL)[e] = outv;
}

// ------------------------------- main kernel --------------------------------
// One 16-sample tile per block.  4 waves: (left,K0) (left,K1) (right,K0)
// (right,K1); each runs 32 K-steps of v_wmma_f32_16x16x32_bf16 over 4 N-tiles,
// stages its 16x64 f32 partial to LDS, then 16 lanes do the bilinear epilogue.
__global__ __launch_bounds__(128, 1) void mps_main(
    const float* __restrict__ x,
    const float* __restrict__ wsf,
    const uint32_t* __restrict__ WL,
    const uint32_t* __restrict__ WR,
    float* __restrict__ out) {
  __shared__ float lds[4][16 * 64];

  const int tid  = threadIdx.x;
  const int wave = tid >> 5;          // wave32 on gfx1250
  const int lane = tid & 31;
  const int g    = lane >> 4;
  const int m    = lane & 15;
  const long m0  = (long)blockIdx.x << 4;

  const uint32_t* __restrict__ W = (wave < 2) ? WL : WR;
  const int  kbBase = (wave & 1) << 5;                 // K-blocks 0..31 or 32..63
  const long rowOff = (wave < 2) ? 0 : 2048;           // both 16B-aligned now
  const float* __restrict__ rowPtr = x + (m0 + m) * 4096 + rowOff;

  v8f acc0 = {}, acc1 = {}, acc2 = {}, acc3 = {};
  union Frag { v16bf v; uint32_t u[8]; };

  for (int i = 0; i < 32; ++i) {
    const int kb = kbBase + i;

    // ---- A fragment: 16-bit A 16x32 layout (ISA table).  Lane group g needs
    // K = 8g..8g+7 (VGPR 0-3) and K = 16+8g..16+8g+7 (VGPR 4-7): two aligned
    // float4 pairs.  Native __bf16 casts -> backend emits packed converts.
    const float* pA = rowPtr + kb * 32 + (g << 3);
    const float4* c0 = reinterpret_cast<const float4*>(pA);        // K 8g..8g+7
    const float4* c1 = reinterpret_cast<const float4*>(pA + 16);   // K 16+8g..
    const float4 f0 = c0[0], f1 = c0[1], f2 = c1[0], f3 = c1[1];
    __builtin_prefetch(pA + 512, 0, 1);   // stream hint: ~16 K-steps ahead
    Frag a;
    a.v = (v16bf){(__bf16)f0.x, (__bf16)f0.y, (__bf16)f0.z, (__bf16)f0.w,
                  (__bf16)f1.x, (__bf16)f1.y, (__bf16)f1.z, (__bf16)f1.w,
                  (__bf16)f2.x, (__bf16)f2.y, (__bf16)f2.z, (__bf16)f2.w,
                  (__bf16)f3.x, (__bf16)f3.y, (__bf16)f3.z, (__bf16)f3.w};

    // ---- B fragments: pre-swizzled, two aligned uint4 loads per N-tile
    const uint32_t* bp = W + (size_t)kb * 1024 + lane * 8;
    #pragma unroll
    for (int nt = 0; nt < 4; ++nt) {
      Frag b;
      const uint4 lo = *reinterpret_cast<const uint4*>(bp + nt * 256);
      const uint4 hi = *reinterpret_cast<const uint4*>(bp + nt * 256 + 4);
      b.u[0] = lo.x; b.u[1] = lo.y; b.u[2] = lo.z; b.u[3] = lo.w;
      b.u[4] = hi.x; b.u[5] = hi.y; b.u[6] = hi.z; b.u[7] = hi.w;
      v8f& acc = (nt == 0) ? acc0 : (nt == 1) ? acc1 : (nt == 2) ? acc2 : acc3;
      acc = __builtin_amdgcn_wmma_f32_16x16x32_bf16(
          /*neg_a=*/false, a.v, /*neg_b=*/false, b.v,
          /*c_mod=*/(short)0, acc, /*reuse_a=*/false, /*reuse_b=*/false);
    }
  }

  // ---- stage accumulators: C/D layout (VGPR r, lane) -> M = r + 8g, N = 16nt+m
  float* dst = &lds[wave][0];
  #pragma unroll
  for (int nt = 0; nt < 4; ++nt) {
    const v8f& acc = (nt == 0) ? acc0 : (nt == 1) ? acc1 : (nt == 2) ? acc2 : acc3;
    #pragma unroll
    for (int r = 0; r < 8; ++r)
      dst[(r + (g << 3)) * 64 + nt * 16 + m] = acc[r];
  }
  __syncthreads();

  // ---- fused epilogue: out[k] = sum_b (lvec.Lmat[:,b]) * (rvec.Rmat[:,b])
  if (tid < 16) {
    const long k = m0 + tid;
    const float* xr = x + k * 4096;
    const float x0 = xr[0],    x1 = xr[1];        // site 0
    const float y0 = xr[4094], y1 = xr[4095];     // site 2047
    float lv[8], rv[8];
    #pragma unroll
    for (int a8 = 0; a8 < 8; ++a8) {
      lv[a8] = wsf[2 * a8] * x0 + wsf[2 * a8 + 1] * x1;
      rv[a8] = wsf[16 + 2 * a8] * y0 + wsf[16 + 2 * a8 + 1] * y1;
    }
    const float* L0 = &lds[0][tid * 64];
    const float* L1 = &lds[1][tid * 64];
    const float* R0 = &lds[2][tid * 64];
    const float* R1 = &lds[3][tid * 64];
    float o = 0.f;
    #pragma unroll
    for (int bb = 0; bb < 8; ++bb) {
      float lsum = 0.f, rsum = 0.f;
      #pragma unroll
      for (int aa = 0; aa < 8; ++aa) {
        const int n = (aa << 3) + bb;
        lsum += lv[aa] * (L0[n] + L1[n]);
        rsum += rv[aa] * (R0[n] + R1[n]);
      }
      o += lsum * rsum;
    }
    out[k] = o;
  }
}

// ------------------------------- launcher -----------------------------------
extern "C" void kernel_launch(void* const* d_in, const int* in_sizes, int n_in,
                              void* d_out, int out_size, void* d_ws, size_t ws_size,
                              hipStream_t stream) {
  const float* x       = (const float*)d_in[0];
  const float* left_b  = (const float*)d_in[1];
  const float* left    = (const float*)d_in[2];
  const float* right   = (const float*)d_in[3];
  const float* right_b = (const float*)d_in[4];
  float*    out = (float*)d_out;
  float*    wsf = (float*)d_ws;
  uint32_t* WL  = (uint32_t*)((char*)d_ws + 256);
  uint32_t* WR  = WL + 65536;

  const int B = in_sizes[0] / 4096;   // 8192

  mps_prep0  <<<1,   64,  0, stream>>>(left_b, left, right, right_b, wsf);
  mps_build_w<<<512, 256, 0, stream>>>(left, right, wsf, WL, WR);
  mps_main   <<<B / 16, 128, 0, stream>>>(x, wsf, WL, WR, out);
}
